// LSTM_AE_77610059039026
// MI455X (gfx1250) — matrix operand order repaired and checked by reading, not compiled, not run on hardware
//
#include <hip/hip_runtime.h>
#include <hip/hip_bf16.h>

typedef __bf16 bf16_t;
typedef __attribute__((ext_vector_type(16))) __bf16 v16bf;
typedef __attribute__((ext_vector_type(8)))  __bf16 v8bf;
typedef __attribute__((ext_vector_type(8)))  float  v8f;

#define HID   128
#define GATES 512
#define TSEQ  1024
#define BATCH 256
#define MROWS 16
#define GSTR  516   // padded f32 stride for gate buffer (bank-conflict break)

union Frag16 { v16bf v; v8bf h[2]; };

// A-matrix fragment (16x32 bf16, MxK): lane holds row m = lane&15.
// lanes 0-15: K = {k0+0..7, k0+16..23}; lanes 16-31: K = {k0+8..15, k0+24..31}.
__device__ __forceinline__ v16bf load_a_frag(const bf16_t* base, int ldA, int k0, int lane) {
    int m  = lane & 15;
    int hi = lane >> 4;
    const bf16_t* p = base + m * ldA + k0 + hi * 8;
    Frag16 f;
    f.h[0] = *(const v8bf*)(p);
    f.h[1] = *(const v8bf*)(p + 16);
    return f.v;
}

// B-matrix fragment (32x16 bf16, KxN) from W^T stored [K][N] row-major:
// lane holds column n = n0 + (lane&15); K = {k0+8h+0..7, k0+16+8h+0..7}, h = lane>>4.
__device__ __forceinline__ v16bf load_b_frag(const bf16_t* wt, int ldB, int k0, int n0, int lane) {
    int hi = lane >> 4;
    int n  = n0 + (lane & 15);
    Frag16 f;
#pragma unroll
    for (int s = 0; s < 16; ++s) {
        int k = k0 + hi * 8 + (s & 7) + (s >> 3) * 16;
        f.v[s] = wt[(size_t)k * ldB + n];
    }
    return f.v;
}

__device__ __forceinline__ float fast_sigmoid(float x) {
    return 1.0f / (1.0f + __expf(-x));
}
__device__ __forceinline__ float fast_tanh(float x) {
    float ax = fabsf(x);
    float e  = __expf(-2.0f * ax);
    float r  = (1.0f - e) / (1.0f + e);
    return (x < 0.0f) ? -r : r;
}

// ---------------------------------------------------------------------------
// Prep: dst[k*N + n] = (bf16) src[n*K + k]   (W [N][K] row-major -> W^T [K][N])
// ---------------------------------------------------------------------------
__global__ void transpose_cvt_kernel(const float* __restrict__ src,
                                     bf16_t* __restrict__ dst, int N, int K) {
    int idx = blockIdx.x * blockDim.x + threadIdx.x;
    if (idx < N * K) {
        int k = idx / N;
        int n = idx % N;
        dst[idx] = (bf16_t)src[n * K + k];
    }
}

// ---------------------------------------------------------------------------
// Persistent LSTM layer: one workgroup owns 16 batch rows for all T steps.
// gates = h @ WhhT + x_t @ WihT + b ; c = f*c + i*g ; h = o*tanh(c)
// XMODE: 0 = no input (zeros), 1 = f32 input Din=32, 2 = bf16 input Din=128
// ---------------------------------------------------------------------------
template<int XMODE, bool HAS_OUT, bool INIT_STATE, bool OUT_STATE>
__global__ __launch_bounds__(256)
void lstm_layer_kernel(const void* __restrict__ xin,
                       const bf16_t* __restrict__ WihT,   // [Din][512] bf16
                       const bf16_t* __restrict__ WhhT,   // [128][512] bf16
                       const float*  __restrict__ bias,   // [512]
                       const float*  __restrict__ h0,     // [256][128] (INIT_STATE)
                       const float*  __restrict__ c0,     // [256][128] (INIT_STATE)
                       bf16_t* __restrict__ out_seq,      // [256][1024][128] (HAS_OUT)
                       float*  __restrict__ hT,           // [256][128] (OUT_STATE)
                       float*  __restrict__ cT)           // [256][128] (OUT_STATE)
{
    constexpr int DIN = (XMODE == 2) ? HID : (XMODE == 1 ? 32 : 0);
    constexpr int XK  = DIN / 32;                 // input-matmul k-chunks

    __shared__ bf16_t sWhhT[HID * GATES];                       // 128 KB
    __shared__ bf16_t sWihT[(XMODE ? DIN : 8) * GATES];         // 128/32 KB
    __shared__ bf16_t sH[MROWS * HID];                          //   4 KB
    __shared__ bf16_t sX[XMODE ? MROWS * HID : 8];              //   4 KB
    __shared__ float  sG[MROWS * GSTR];                         //  33 KB

    const int tid  = threadIdx.x;
    const int lane = tid & 31;
    const int wv   = tid >> 5;                    // 8 waves (wave32)
    const int b0   = blockIdx.x * MROWS;          // batch-row tile

    // ---- one-time staging of weights into LDS (contiguous 16B chunks) ----
    {
        const v8bf* src = (const v8bf*)WhhT;
        v8bf*       dst = (v8bf*)sWhhT;
        for (int i = tid; i < HID * GATES / 8; i += 256) dst[i] = src[i];
        if (XMODE) {
            const v8bf* s2 = (const v8bf*)WihT;
            v8bf*       d2 = (v8bf*)sWihT;
            for (int i = tid; i < DIN * GATES / 8; i += 256) d2[i] = s2[i];
        }
    }
    // ---- init hidden state (LDS, bf16) and cell state (registers, f32) ----
    const int m_e = tid >> 4;                     // elementwise row 0..15
    const int j0  = (tid & 15) * 8;               // elementwise col base
    for (int i = tid; i < MROWS * HID; i += 256) {
        float hv = INIT_STATE ? h0[(b0 + (i >> 7)) * HID + (i & 127)] : 0.0f;
        sH[i] = (bf16_t)hv;
    }
    float creg[8];
#pragma unroll
    for (int p = 0; p < 8; ++p)
        creg[p] = INIT_STATE ? c0[(b0 + m_e) * HID + j0 + p] : 0.0f;

    // ---- per-lane constants for the matmul phase ----
    const int   hi  = lane >> 4;
    const int   nl  = lane & 15;
    const int   n0w = wv * 64;                    // this wave's 4 N-tiles
    float bv[4];
#pragma unroll
    for (int nt = 0; nt < 4; ++nt) bv[nt] = bias[n0w + nt * 16 + nl];

    // ---- per-thread streaming pointers (advance by one timestep each iter) ----
    const float*  xs1 = nullptr; const float* xs2 = nullptr;   // XMODE==1
    const bf16_t* xs3 = nullptr;                                // XMODE==2
    int d1i = 0, d2i = 0;
    if (XMODE == 1) {
        const int m1 = tid >> 5, k1 = tid & 31;   // element 1: rows 0..7
        xs1 = (const float*)xin + (size_t)(b0 + m1) * TSEQ * 32 + k1;
        xs2 = xs1 + (size_t)8 * TSEQ * 32;        // element 2: rows 8..15
        d1i = m1 * HID + k1;
        d2i = (m1 + 8) * HID + k1;
    } else if (XMODE == 2) {
        const int mx = tid >> 4, c8 = tid & 15;   // one 16B chunk per thread
        xs3 = (const bf16_t*)xin + (size_t)(b0 + mx) * TSEQ * HID + c8 * 8;
    }
    bf16_t* outp = nullptr;
    if (HAS_OUT) outp = out_seq + (size_t)(b0 + m_e) * TSEQ * HID + j0;

    __syncthreads();

#pragma unroll 1
    for (int t = 0; t < TSEQ; ++t) {
        // ---- stage x_t into LDS ----
        if (XMODE == 1) {
            sX[d1i] = (bf16_t)*xs1;
            sX[d2i] = (bf16_t)*xs2;
            __builtin_prefetch(xs1 + 32, 0, 1);
            __builtin_prefetch(xs2 + 32, 0, 1);
            xs1 += 32; xs2 += 32;
        } else if (XMODE == 2) {
            ((v8bf*)sX)[tid] = *(const v8bf*)xs3;
            __builtin_prefetch(xs3 + HID, 0, 1);
            xs3 += HID;
        }
        __syncthreads();

        // ---- matmul phase: gates_pre -> sG ----
        v16bf ah[4];
#pragma unroll
        for (int kc = 0; kc < 4; ++kc) ah[kc] = load_a_frag(sH, HID, kc * 32, lane);

#pragma unroll
        for (int nt = 0; nt < 4; ++nt) {
            const int n0 = n0w + nt * 16;
            v8f acc = {};
#pragma unroll
            for (int kc = 0; kc < 4; ++kc) {
                v16bf bf = load_b_frag(sWhhT, GATES, kc * 32, n0, lane);
                acc = __builtin_amdgcn_wmma_f32_16x16x32_bf16(
                          false, ah[kc], false, bf, (short)0, acc, false, false);
            }
#pragma unroll
            for (int kc = 0; kc < XK; ++kc) {
                v16bf af = load_a_frag(sX, HID, kc * 32, lane);
                v16bf bf = load_b_frag(sWihT, GATES, kc * 32, n0, lane);
                acc = __builtin_amdgcn_wmma_f32_16x16x32_bf16(
                          false, af, false, bf, (short)0, acc, false, false);
            }
            const int nn = n0 + nl;
#pragma unroll
            for (int r = 0; r < 8; ++r)            // C/D: M = r + 8*hi, N = lane&15
                sG[(r + hi * 8) * GSTR + nn] = acc[r] + bv[nt];
        }
        __syncthreads();

        // ---- elementwise LSTM cell update (one row-slice per thread) ----
        {
            const float* grow = &sG[m_e * GSTR];
            v8bf hvec;
#pragma unroll
            for (int p = 0; p < 8; ++p) {
                float gi = grow[  0 + j0 + p];
                float gf = grow[128 + j0 + p];
                float gg = grow[256 + j0 + p];
                float go = grow[384 + j0 + p];
                float iv = fast_sigmoid(gi);
                float fv = fast_sigmoid(gf);
                float gv = fast_tanh(gg);
                float ov = fast_sigmoid(go);
                float cc = fv * creg[p] + iv * gv;
                creg[p]  = cc;
                hvec[p]  = (bf16_t)(ov * fast_tanh(cc));
            }
            *(v8bf*)(&sH[m_e * HID + j0]) = hvec;   // 16B LDS store
            if (HAS_OUT) {
                *(v8bf*)outp = hvec;                // 16B global store
                outp += HID;
            }
        }
        __syncthreads();
    }

    // ---- final states for decoder init ----
    if (OUT_STATE) {
#pragma unroll
        for (int p = 0; p < 8; ++p) {
            cT[(b0 + m_e) * HID + j0 + p] = creg[p];
            hT[(b0 + m_e) * HID + j0 + p] = (float)sH[m_e * HID + j0 + p];
        }
    }
}

// ---------------------------------------------------------------------------
// Output projection: out[M][32] = A_bf16[M][128] @ WT_bf16[128][32] + b
// block = 256 threads (8 waves), each block handles 128 rows.
// ---------------------------------------------------------------------------
__global__ __launch_bounds__(256)
void proj_kernel(const bf16_t* __restrict__ A,    // [M][128] bf16
                 const bf16_t* __restrict__ WT,   // [128][32] bf16
                 const float*  __restrict__ bias, // [32]
                 float* __restrict__ out)         // [M][32] f32
{
    __shared__ bf16_t sA[128 * HID];  // 32 KB
    __shared__ bf16_t sB[HID * 32];   //  8 KB

    const int tid  = threadIdx.x;
    const int lane = tid & 31;
    const int wv   = tid >> 5;
    const size_t m0 = (size_t)blockIdx.x * 128;

    for (int i = tid; i < 128 * HID / 8; i += 256)
        ((v8bf*)sA)[i] = ((const v8bf*)(A + m0 * HID))[i];
    for (int i = tid; i < HID * 32 / 8; i += 256)
        ((v8bf*)sB)[i] = ((const v8bf*)WT)[i];
    __syncthreads();

    const bf16_t* aw = sA + wv * 16 * HID;
#pragma unroll
    for (int nt = 0; nt < 2; ++nt) {
        v8f acc = {};
#pragma unroll
        for (int kc = 0; kc < 4; ++kc) {
            v16bf af = load_a_frag(aw, HID, kc * 32, lane);
            v16bf bf = load_b_frag(sB, 32, kc * 32, nt * 16, lane);
            acc = __builtin_amdgcn_wmma_f32_16x16x32_bf16(
                      false, af, false, bf, (short)0, acc, false, false);
        }
        const int hi = lane >> 4;
        const int nn = nt * 16 + (lane & 15);
        const float bvv = bias[nn];
#pragma unroll
        for (int r = 0; r < 8; ++r) {
            size_t row = m0 + (size_t)wv * 16 + r + hi * 8;
            out[row * 32 + nn] = acc[r] + bvv;
        }
    }
}

// ---------------------------------------------------------------------------
// Host: workspace layout + launch sequence (all on `stream`)
// ---------------------------------------------------------------------------
extern "C" void kernel_launch(void* const* d_in, const int* in_sizes, int n_in,
                              void* d_out, int out_size, void* d_ws, size_t ws_size,
                              hipStream_t stream) {
    (void)in_sizes; (void)n_in; (void)out_size; (void)ws_size;

    const float* x        = (const float*)d_in[0];
    const float* enc_Wih0 = (const float*)d_in[1];
    const float* enc_Whh0 = (const float*)d_in[2];
    const float* enc_b0   = (const float*)d_in[3];
    const float* enc_Wih1 = (const float*)d_in[4];
    const float* enc_Whh1 = (const float*)d_in[5];
    const float* enc_b1   = (const float*)d_in[6];
    /* dec_Wih0 (d_in[7]) unused: decoder input is zeros */
    const float* dec_Whh0 = (const float*)d_in[8];
    const float* dec_b0   = (const float*)d_in[9];
    const float* dec_Wih1 = (const float*)d_in[10];
    const float* dec_Whh1 = (const float*)d_in[11];
    const float* dec_b1   = (const float*)d_in[12];
    const float* out_W    = (const float*)d_in[13];
    const float* out_b    = (const float*)d_in[14];

    char* ws = (char*)d_ws;
    const size_t SEQ_B   = (size_t)BATCH * TSEQ * HID * sizeof(bf16_t); // 64 MB
    const size_t WHH_B   = (size_t)HID * GATES * sizeof(bf16_t);        // 128 KB
    const size_t WIH32_B = (size_t)32 * GATES * sizeof(bf16_t);
    const size_t OUTW_B  = (size_t)HID * 32 * sizeof(bf16_t);
    const size_t ST_B    = (size_t)BATCH * HID * sizeof(float);

    size_t off = 0;
    bf16_t* e0       = (bf16_t*)(ws + off); off += SEQ_B;
    bf16_t* d0buf    = (bf16_t*)(ws + off); off += SEQ_B;
    bf16_t* d1buf    = (bf16_t*)(ws + off); off += SEQ_B;
    bf16_t* WhhT_e0  = (bf16_t*)(ws + off); off += WHH_B;
    bf16_t* WhhT_e1  = (bf16_t*)(ws + off); off += WHH_B;
    bf16_t* WhhT_d0  = (bf16_t*)(ws + off); off += WHH_B;
    bf16_t* WhhT_d1  = (bf16_t*)(ws + off); off += WHH_B;
    bf16_t* WihT_e0  = (bf16_t*)(ws + off); off += WIH32_B;
    bf16_t* WihT_e1  = (bf16_t*)(ws + off); off += WHH_B;
    bf16_t* WihT_d1  = (bf16_t*)(ws + off); off += WHH_B;
    bf16_t* outWT    = (bf16_t*)(ws + off); off += OUTW_B;
    float*  h_e0     = (float*)(ws + off);  off += ST_B;
    float*  c_e0     = (float*)(ws + off);  off += ST_B;
    float*  h_e1     = (float*)(ws + off);  off += ST_B;
    float*  c_e1     = (float*)(ws + off);  off += ST_B;

    // ---- prep: transpose + convert all weights to bf16 W^T ----
    auto tc = [&](const float* src, bf16_t* dst, int N, int K) {
        int n = N * K;
        transpose_cvt_kernel<<<(n + 255) / 256, 256, 0, stream>>>(src, dst, N, K);
    };
    tc(enc_Whh0, WhhT_e0, GATES, HID);
    tc(enc_Whh1, WhhT_e1, GATES, HID);
    tc(dec_Whh0, WhhT_d0, GATES, HID);
    tc(dec_Whh1, WhhT_d1, GATES, HID);
    tc(enc_Wih0, WihT_e0, GATES, 32);
    tc(enc_Wih1, WihT_e1, GATES, HID);
    tc(dec_Wih1, WihT_d1, GATES, HID);
    tc(out_W,    outWT,   32,    HID);

    // ---- encoder layer 0: x (f32, Din=32) -> e0, final (h_e0, c_e0) ----
    lstm_layer_kernel<1, true, false, true><<<BATCH / MROWS, 256, 0, stream>>>(
        x, WihT_e0, WhhT_e0, enc_b0, nullptr, nullptr, e0, h_e0, c_e0);

    // ---- encoder layer 1: e0 (bf16, Din=128) -> final (h_e1, c_e1) only ----
    lstm_layer_kernel<2, false, false, true><<<BATCH / MROWS, 256, 0, stream>>>(
        e0, WihT_e1, WhhT_e1, enc_b1, nullptr, nullptr, nullptr, h_e1, c_e1);

    // ---- decoder layer 0: zero input, init (h_e0, c_e0) -> d0 ----
    lstm_layer_kernel<0, true, true, false><<<BATCH / MROWS, 256, 0, stream>>>(
        nullptr, nullptr, WhhT_d0, dec_b0, h_e0, c_e0, d0buf, nullptr, nullptr);

    // ---- decoder layer 1: d0 (bf16), init (h_e1, c_e1) -> d1 ----
    lstm_layer_kernel<2, true, true, false><<<BATCH / MROWS, 256, 0, stream>>>(
        d0buf, WihT_d1, WhhT_d1, dec_b1, h_e1, c_e1, d1buf, nullptr, nullptr);

    // ---- projection: d1 @ out_W^T + out_b -> d_out [B*T][32] f32 ----
    proj_kernel<<<(BATCH * TSEQ) / 128, 256, 0, stream>>>(
        d1buf, outWT, out_b, (float*)d_out);
}
